// AutoEncoderLSTM_otoi_56581899158119
// MI455X (gfx1250) — matrix operand order repaired and checked by reading, not compiled, not run on hardware
//
#include <hip/hip_runtime.h>
#include <hip/hip_bf16.h>
#include <stdint.h>

// ---------------------------------------------------------------------------
// LSTM autoencoder for MI455X (gfx1250, wave32, WMMA bf16 16x16x32).
// B=128, T=512, F=64, H=512.  8 persistent workgroups (one 16-row batch slice
// each), 16 waves per WG; h in LDS, c in registers, gates in WMMA accumulators.
// Decoder uses pre-summed (Wih+Whh) since x==h after step 1.
// Round 2: unroll-1 k-loops (kill VGPR spills), input pre-packed to bf16
// A-layout by a prep kernel (no f32->bf16 VALU in the hot loop).
// ---------------------------------------------------------------------------

typedef __bf16 v16bf __attribute__((ext_vector_type(16)));
typedef float  v8f   __attribute__((ext_vector_type(8)));

#define BATCH 128
#define TLEN  512
#define FDIM  64
#define HDIM  512
#define HSTRIDE 520   // padded LDS row stride in bf16 elems (1040 B, 16B aligned)

union BfVec { uint4 u[2]; unsigned short s[16]; v16bf v; };

__device__ __forceinline__ unsigned short f2bf(float f) {
  union { float f; unsigned u; } x; x.f = f;
  unsigned r = x.u + 0x7FFFu + ((x.u >> 16) & 1u);   // round-to-nearest-even
  return (unsigned short)(r >> 16);
}
__device__ __forceinline__ float sigf(float x)     { return 1.0f / (1.0f + __expf(-x)); }
__device__ __forceinline__ float tanhfast(float x) { return 1.0f - 2.0f / (__expf(2.0f * x) + 1.0f); }

__device__ __forceinline__ v8f wmma_bf16(const BfVec& a, const BfVec& b, v8f c) {
  return __builtin_amdgcn_wmma_f32_16x16x32_bf16(false, a.v, false, b.v, (short)0, c, false, false);
}

// Packed B-operand tile: ((nt*KT + kt)*32 + lane)*16 bf16, 32 B per lane.
__device__ __forceinline__ BfVec loadB(const uint4* __restrict__ pack,
                                       int nt, int kt, int KT, int lane) {
  BfVec b;
  const uint4* q = pack + (((size_t)nt * KT + kt) * 32 + lane) * 2;
  b.u[0] = q[0]; b.u[1] = q[1];
  return b;
}

// ---------------------------------------------------------------------------
// Prep: pack fp32 weight [N,K] into bf16 WMMA B-layout tiles.
// Optional src2 is added (for Wih_dec + Whh_dec pre-sum).
// ---------------------------------------------------------------------------
__global__ void pack_w_bf16(const float* __restrict__ src,
                            const float* __restrict__ src2,
                            unsigned short* __restrict__ dst,
                            int K, int total) {
  int tid = blockIdx.x * blockDim.x + threadIdx.x;
  if (tid >= total) return;
  int KT   = K >> 5;
  int i    = tid & 15;
  int lane = (tid >> 4) & 31;
  int kt   = (tid >> 9) % KT;
  int nt   = tid / (512 * KT);
  int n = nt * 16 + (lane & 15);
  int k = kt * 32 + (lane >> 4) * 8 + (i & 7) + (i >> 3) * 16;
  float v = src[(size_t)n * K + k];
  if (src2) v += src2[(size_t)n * K + k];
  dst[tid] = f2bf(v);
}

// ---------------------------------------------------------------------------
// Prep: pack input_seq [B,T,F] fp32 into bf16 WMMA A-layout blocks:
// dst[(((bt*T + t)*2 + kt)*512) + lane*16 + i],   bt = batch tile of 16 rows.
// A layout: lane<16 row=lane, K runs {k0..k0+7, k0+16..k0+23}; lanes 16-31
// same rows, runs {k0+8..15, k0+24..31}.
// ---------------------------------------------------------------------------
__global__ void pack_x_bf16(const float* __restrict__ src,
                            unsigned short* __restrict__ dst, int total) {
  int tid = blockIdx.x * blockDim.x + threadIdx.x;
  if (tid >= total) return;
  int i    = tid & 15;
  int lane = (tid >> 4) & 31;
  int kt   = (tid >> 9) & 1;
  int t    = (tid >> 10) & (TLEN - 1);
  int bt   = tid >> 19;
  int b    = bt * 16 + (lane & 15);
  int k    = kt * 32 + (lane >> 4) * 8 + (i & 7) + (i >> 3) * 16;
  dst[tid] = f2bf(src[((size_t)b * TLEN + t) * FDIM + k]);
}

__global__ void add_bias(const float* __restrict__ a, const float* __restrict__ b,
                         float* __restrict__ o, int n) {
  int i = blockIdx.x * blockDim.x + threadIdx.x;
  if (i < n) o[i] = a[i] + b[i];
}

// ---------------------------------------------------------------------------
// Main persistent kernel. grid = 8 blocks (batch slices), block = 512 threads.
// ---------------------------------------------------------------------------
__global__ void __launch_bounds__(512)
lstm_ae_main(const uint4* __restrict__ pX,
             const uint4* __restrict__ pWihE, const uint4* __restrict__ pWhhE,
             const uint4* __restrict__ pWihD, const uint4* __restrict__ pWsum,
             const uint4* __restrict__ pWlin,
             const float* __restrict__ be, const float* __restrict__ bd,
             const float* __restrict__ blin,
             float* __restrict__ out)
{
  __shared__ unsigned short hbuf[16 * HSTRIDE];   // h_t, bf16, 16 rows x 512 (+pad)

  const int tid   = threadIdx.x;
  const int wave  = tid >> 5;
  const int lane  = tid & 31;
  const int row16 = lane & 15;
  const int rhal  = lane >> 4;        // 0: lanes 0-15, 1: lanes 16-31
  const int b0    = blockIdx.x * 16;

  for (int i = tid; i < 16 * HSTRIDE; i += 512) hbuf[i] = 0;
  __syncthreads();

  // Wave w owns hidden band [32w, 32w+32): 8 gate N-tiles (4 gates x 2 subtiles)
  int ntb[8];
#pragma unroll
  for (int i = 0; i < 8; ++i) ntb[i] = 32 * (i >> 1) + 2 * wave + (i & 1);

  float eb[8], db[8];
#pragma unroll
  for (int i = 0; i < 8; ++i) {
    eb[i] = be[ntb[i] * 16 + row16];
    db[i] = bd[ntb[i] * 16 + row16];
  }
  const float pb = (wave < 4) ? blin[wave * 16 + row16] : 0.0f;

  float c[16];
#pragma unroll
  for (int j = 0; j < 16; ++j) c[j] = 0.0f;

  // A-operand from LDS h
  auto loadA = [&](int kt) {
    BfVec a;
    const uint4* p = (const uint4*)(&hbuf[row16 * HSTRIDE + kt * 32 + rhal * 8]);
    a.u[0] = p[0]; a.u[1] = p[2];
    return a;
  };

  // Gate nonlinearity + c/h update, h -> LDS (bf16); optionally dump f32 h.
  auto update = [&](v8f* acc, bool writeLast) {
#pragma unroll
    for (int tp = 0; tp < 2; ++tp) {
#pragma unroll
      for (int v = 0; v < 8; ++v) {
        float iv = acc[0 + tp][v], fv = acc[2 + tp][v];
        float gv = acc[4 + tp][v], ov = acc[6 + tp][v];
        float cn = sigf(fv) * c[tp * 8 + v] + sigf(iv) * tanhfast(gv);
        float hn = sigf(ov) * tanhfast(cn);
        c[tp * 8 + v] = cn;
        const int row = v + 8 * rhal;
        const int hid = 32 * wave + 16 * tp + row16;
        hbuf[row * HSTRIDE + hid] = f2bf(hn);
        if (writeLast)
          out[(size_t)BATCH * TLEN * FDIM + (size_t)(b0 + row) * HDIM + hid] = hn;
      }
    }
  };

  // ---------------- Encoder: 512 steps ----------------
  for (int t = 0; t < TLEN; ++t) {
    v8f acc[8];
#pragma unroll
    for (int i = 0; i < 8; ++i) {
      v8f z;
#pragma unroll
      for (int e = 0; e < 8; ++e) z[e] = eb[i];
      acc[i] = z;
    }
    // x_t part (K = 64): pre-packed bf16 A-layout, 2x b128 per k-tile
#pragma unroll 1
    for (int kt = 0; kt < 2; ++kt) {
      BfVec a;
      const uint4* xq = pX + (((size_t)blockIdx.x * TLEN + t) * 2 + kt) * 64 + lane * 2;
      a.u[0] = xq[0]; a.u[1] = xq[1];
#pragma unroll
      for (int i = 0; i < 8; ++i)
        acc[i] = wmma_bf16(a, loadB(pWihE, ntb[i], kt, 2, lane), acc[i]);
    }
    // h part (K = 512)
#pragma unroll 1
    for (int kt = 0; kt < 16; ++kt) {
      BfVec a = loadA(kt);
#pragma unroll
      for (int i = 0; i < 8; ++i)
        acc[i] = wmma_bf16(a, loadB(pWhhE, ntb[i], kt, 16, lane), acc[i]);
    }
    __syncthreads();
    update(acc, false);
    __syncthreads();
  }

  // ---------------- Decoder: 512 steps (x == h after step 1) ----------------
#pragma unroll
  for (int j = 0; j < 16; ++j) c[j] = 0.0f;

  for (int t = 0; t < TLEN; ++t) {
    const uint4* W = (t == 0) ? pWihD : pWsum;  // t==0: x=h_enc, h=0
    v8f acc[8];
#pragma unroll
    for (int i = 0; i < 8; ++i) {
      v8f z;
#pragma unroll
      for (int e = 0; e < 8; ++e) z[e] = db[i];
      acc[i] = z;
    }
    v8f pacc;
#pragma unroll
    for (int e = 0; e < 8; ++e) pacc[e] = pb;
    const bool doproj = (wave < 4) && (t > 0);  // project h_t (valid outputs t>=1)

#pragma unroll 1
    for (int kt = 0; kt < 16; ++kt) {
      BfVec a = loadA(kt);
#pragma unroll
      for (int i = 0; i < 8; ++i)
        acc[i] = wmma_bf16(a, loadB(W, ntb[i], kt, 16, lane), acc[i]);
      if (doproj)
        pacc = wmma_bf16(a, loadB(pWlin, wave, kt, 16, lane), pacc);
    }
    if (doproj) {
      const int tout = TLEN - t;          // h_t stored at time index 512 - t
      const int col  = wave * 16 + row16;
#pragma unroll
      for (int v = 0; v < 8; ++v)
        out[((size_t)(b0 + v + 8 * rhal) * TLEN + tout) * FDIM + col] = pacc[v];
    }
    __syncthreads();
    update(acc, t == TLEN - 1);           // last step also writes x_last (f32)
    __syncthreads();
  }

  // Final projection of h_512 -> output time index 0
  if (wave < 4) {
    v8f pacc;
#pragma unroll
    for (int e = 0; e < 8; ++e) pacc[e] = pb;
#pragma unroll 1
    for (int kt = 0; kt < 16; ++kt) {
      BfVec a = loadA(kt);
      pacc = wmma_bf16(a, loadB(pWlin, wave, kt, 16, lane), pacc);
    }
    const int col = wave * 16 + row16;
#pragma unroll
    for (int v = 0; v < 8; ++v)
      out[((size_t)(b0 + v + 8 * rhal) * TLEN + 0) * FDIM + col] = pacc[v];
  }
}

// ---------------------------------------------------------------------------
// Launcher
// ---------------------------------------------------------------------------
extern "C" void kernel_launch(void* const* d_in, const int* in_sizes, int n_in,
                              void* d_out, int out_size, void* d_ws, size_t ws_size,
                              hipStream_t stream) {
  const float* input  = (const float*)d_in[0];
  const float* WihE   = (const float*)d_in[1];
  const float* WhhE   = (const float*)d_in[2];
  const float* bihE   = (const float*)d_in[3];
  const float* bhhE   = (const float*)d_in[4];
  const float* WihD   = (const float*)d_in[5];
  const float* WhhD   = (const float*)d_in[6];
  const float* bihD   = (const float*)d_in[7];
  const float* bhhD   = (const float*)d_in[8];
  const float* Wlin   = (const float*)d_in[9];
  const float* blin   = (const float*)d_in[10];
  float* out = (float*)d_out;

  // Workspace layout (bytes)
  char* w = (char*)d_ws;
  unsigned short* pWihE = (unsigned short*)(w + 0);          //  2048*64  bf16
  unsigned short* pWhhE = (unsigned short*)(w + 262144);     //  2048*512 bf16
  unsigned short* pWihD = (unsigned short*)(w + 2359296);    //  2048*512 bf16
  unsigned short* pWsum = (unsigned short*)(w + 4456448);    //  2048*512 bf16
  unsigned short* pWlin = (unsigned short*)(w + 6553600);    //  64*512   bf16
  float*          be    = (float*)(w + 6619136);             //  2048 f32
  float*          bd    = (float*)(w + 6627328);             //  2048 f32
  unsigned short* pX    = (unsigned short*)(w + 6635520);    //  128*512*64 bf16

  // Pack weights to bf16 WMMA B-layout (decoder matrices pre-summed)
  pack_w_bf16<<<(2048 * 64 + 255) / 256, 256, 0, stream>>>(WihE, nullptr, pWihE, 64, 2048 * 64);
  pack_w_bf16<<<(2048 * 512 + 255) / 256, 256, 0, stream>>>(WhhE, nullptr, pWhhE, 512, 2048 * 512);
  pack_w_bf16<<<(2048 * 512 + 255) / 256, 256, 0, stream>>>(WihD, nullptr, pWihD, 512, 2048 * 512);
  pack_w_bf16<<<(2048 * 512 + 255) / 256, 256, 0, stream>>>(WihD, WhhD,   pWsum, 512, 2048 * 512);
  pack_w_bf16<<<(64 * 512 + 255) / 256, 256, 0, stream>>>(Wlin, nullptr, pWlin, 512, 64 * 512);
  add_bias<<<8, 256, 0, stream>>>(bihE, bhhE, be, 2048);
  add_bias<<<8, 256, 0, stream>>>(bihD, bhhD, bd, 2048);
  // Pack input to bf16 WMMA A-layout
  pack_x_bf16<<<(BATCH * TLEN * FDIM + 255) / 256, 256, 0, stream>>>(input, pX, BATCH * TLEN * FDIM);

  // Persistent LSTM: 8 batch slices x 16 waves, no cross-WG sync needed.
  lstm_ae_main<<<8, 512, 0, stream>>>((const uint4*)pX,
                                      (const uint4*)pWihE, (const uint4*)pWhhE,
                                      (const uint4*)pWihD, (const uint4*)pWsum,
                                      (const uint4*)pWlin,
                                      be, bd, blin, out);
}
